// CnUpdateLayer_34565896798954
// MI455X (gfx1250) — matrix operand order repaired and checked by reading, not compiled, not run on hardware
//
#include <hip/hip_runtime.h>
#include <math.h>

typedef __attribute__((ext_vector_type(16))) _Float16 v16h;
typedef __attribute__((ext_vector_type(8)))  _Float16 v8h;
typedef __attribute__((ext_vector_type(4)))  _Float16 v4h;
typedef __attribute__((ext_vector_type(8)))  float    v8f;
typedef __attribute__((ext_vector_type(2)))  float    v2f;

#ifndef __has_builtin
#define __has_builtin(x) 0
#endif

#if __has_builtin(__builtin_amdgcn_wmma_f32_16x16x4_f32)
#define HAVE_WMMA_F32X4 1
#else
#define HAVE_WMMA_F32X4 0
#endif

namespace {
constexpr int kBatch = 128;
constexpr int kOut   = 1024;
constexpr int kIn    = 1024;
constexpr int kBG    = 2;  // batch tiles per wave in the GEMM (256 waves total)

constexpr size_t kXPlane    = (size_t)kBatch * kIn * 2;  // 256 KB (f16)
constexpr size_t kMaskPlane = (size_t)kOut   * kIn * 2;  // 2 MB  (f16)
constexpr size_t kOffLhi = 0;
constexpr size_t kOffLlo = kXPlane;
constexpr size_t kOffT   = 2 * kXPlane;
constexpr size_t kOffM   = 3 * kXPlane;
constexpr size_t kWsNeed = 3 * kXPlane + kMaskPlane;     // ~2.82 MB
}

// ---------------------------------------------------------------------------
// Phase 1a: per-element x transform (done ONCE, not once per o-tile).
//   Lhi+Llo : hi/lo f16 split of log|x|  (~22-bit effective mantissa)
//   T       : 0 (x==0), 1 (x>0), 1025 (x<0)  -> sum = nz + 1024*neg, exact
// ---------------------------------------------------------------------------
__global__ __launch_bounds__(256) void transform_x_kernel(
    const float* __restrict__ x,
    _Float16* __restrict__ lhi, _Float16* __restrict__ llo,
    _Float16* __restrict__ tct)
{
    const int i = (blockIdx.x * 256 + threadIdx.x) * 4;
    const float4 xv4 = *(const float4*)(x + i);
    float xs[4] = { xv4.x, xv4.y, xv4.z, xv4.w };
    v4h hi4, lo4, t4;
#pragma unroll
    for (int j = 0; j < 4; ++j) {
        const float xv = xs[j];
        _Float16 hi = (_Float16)0.0f, lo = (_Float16)0.0f, t = (_Float16)0.0f;
        if (xv != 0.0f) {
            const float L = logf(fabsf(xv));
            hi = (_Float16)L;
            lo = (_Float16)(L - (float)hi);
            t  = (xv < 0.0f) ? (_Float16)1025.0f : (_Float16)1.0f;
        }
        hi4[j] = hi; lo4[j] = lo; t4[j] = t;
    }
    *(v4h*)(lhi + i) = hi4;
    *(v4h*)(llo + i) = lo4;
    *(v4h*)(tct + i) = t4;
}

// Phase 1b: mask f32 -> f16 plane (values 0/1, exact).
__global__ __launch_bounds__(256) void transform_mask_kernel(
    const float* __restrict__ mask, _Float16* __restrict__ mh)
{
    const int i = (blockIdx.x * 256 + threadIdx.x) * 4;  // exact cover: 1M
    const float4 m = *(const float4*)(mask + i);
    v4h h = { (_Float16)m.x, (_Float16)m.y, (_Float16)m.z, (_Float16)m.w };
    *(v4h*)(mh + i) = h;
}

// ---------------------------------------------------------------------------
// Phase 2 helpers: one K=32 chunk of fragments + its 3*kBG WMMAs.
// ---------------------------------------------------------------------------
struct Frags {
    v16h bM;
    v16h aH[kBG];
    v16h aL[kBG];
    v16h aT[kBG];
};

__device__ __forceinline__ v16h load_frag16(const _Float16* p, int k0, int d) {
    const v8h x0 = *(const v8h*)(p + k0);
    const v8h x1 = *(const v8h*)(p + k0 + d);
    return __builtin_shufflevector(x0, x1,
                                   0,1,2,3,4,5,6,7,8,9,10,11,12,13,14,15);
}

__device__ __forceinline__ Frags load_frags(
    const _Float16* pM,
    const _Float16* const* pH, const _Float16* const* pL,
    const _Float16* const* pT, int k0)
{
    Frags f;
    f.bM = load_frag16(pM, k0, 16);   // B: elems 0..7 at +0, 8..15 at +16
#pragma unroll
    for (int t = 0; t < kBG; ++t) {   // A: elems 0..7 at +0, 8..15 at +16
        f.aH[t] = load_frag16(pH[t], k0, 16);
        f.aL[t] = load_frag16(pL[t], k0, 16);
        f.aT[t] = load_frag16(pT[t], k0, 16);
    }
    return f;
}

__device__ __forceinline__ void wmma_step(
    const Frags& f, v8f* accH, v8f* accL, v8f* accT)
{
#pragma unroll
    for (int t = 0; t < kBG; ++t) {
        accH[t] = __builtin_amdgcn_wmma_f32_16x16x32_f16(
                      false, f.aH[t], false, f.bM, (short)0, accH[t], false, false);
        accL[t] = __builtin_amdgcn_wmma_f32_16x16x32_f16(
                      false, f.aL[t], false, f.bM, (short)0, accL[t], false, false);
        accT[t] = __builtin_amdgcn_wmma_f32_16x16x32_f16(
                      false, f.aT[t], false, f.bM, (short)0, accT[t], false, false);
    }
}

// ---------------------------------------------------------------------------
// Phase 2: pure f16 WMMA GEMM, software-pipelined one K-chunk ahead so the
// fragment loads for chunk k+1 fly while the WMMAs for chunk k execute
// (breaks the load->s_wait_loadcnt(0)->wmma serialization).
// ---------------------------------------------------------------------------
__global__ __launch_bounds__(32) void cn_gemm_kernel(
    const _Float16* __restrict__ lhi, const _Float16* __restrict__ llo,
    const _Float16* __restrict__ tct, const _Float16* __restrict__ mh,
    float* __restrict__ out)
{
    const int lane = threadIdx.x;     // 0..31 (wave32)
    const int row  = lane & 15;
    const int hs   = lane >> 4;

    const int o0  = blockIdx.x << 4;           // N tile
    const int bg0 = blockIdx.y * (kBG * 16);   // first M row of this group

    // Loop-invariant per-lane base pointers (only uniform k0 advances).
    const _Float16* pM = mh + (size_t)(o0 + row) * kIn + 8 * hs;
    const _Float16* pH[kBG];
    const _Float16* pL[kBG];
    const _Float16* pT[kBG];
#pragma unroll
    for (int t = 0; t < kBG; ++t) {
        const size_t rb = (size_t)(bg0 + t * 16 + row) * kIn + 8 * hs;
        pH[t] = lhi + rb;
        pL[t] = llo + rb;
        pT[t] = tct + rb;
    }

    v8f accH[kBG], accL[kBG], accT[kBG];
    const v8f vzero = {};
#pragma unroll
    for (int t = 0; t < kBG; ++t) { accH[t] = vzero; accL[t] = vzero; accT[t] = vzero; }

    Frags cur = load_frags(pM, pH, pL, pT, 0);
#pragma unroll 2
    for (int k0 = 32; k0 < kIn; k0 += 32) {
        Frags nxt = load_frags(pM, pH, pL, pT, k0);  // loads in flight...
        wmma_step(cur, accH, accL, accT);            // ...during these WMMAs
        cur = nxt;
    }
    wmma_step(cur, accH, accL, accT);

    // Epilogue: C elem v -> m = v + 8*hs, n = row.
#pragma unroll
    for (int t = 0; t < kBG; ++t) {
#pragma unroll
        for (int v = 0; v < 8; ++v) {
            const float sl = accH[t][v] + accL[t][v];
            const int   tc = (int)(accT[t][v] + 0.5f);
            int neg = tc >> 10;
            int nz  = tc & 1023;
            if (nz == 0 && neg > 0) { nz = 1024; --neg; }  // nz==1024 alias (neg<=nz)
            float val = expf(sl);
            if (neg & 1) val = -val;
            out[(size_t)(bg0 + t * 16 + v + 8 * hs) * kOut + (o0 + row)] =
                (nz > 0) ? val : 0.0f;
        }
    }
}

// ---------------------------------------------------------------------------
// Fallback (fused) kernel, used only if ws_size is too small for the planes.
// ---------------------------------------------------------------------------
__global__ __launch_bounds__(32) void cn_update_fused_kernel(
    const float* __restrict__ x, const float* __restrict__ mask,
    float* __restrict__ out)
{
    const int lane = threadIdx.x;
    const int row  = lane & 15;
    const int hs   = lane >> 4;
    const int o0 = blockIdx.x << 4;
    const int b0 = blockIdx.y << 4;

    const float* __restrict__ xrow = x    + (size_t)(b0 + row) * kIn;
    const float* __restrict__ mrow = mask + (size_t)(o0 + row) * kIn;

    v8f accZ = {}, accS = {};
#if HAVE_WMMA_F32X4
    v8f accLg = {};
#else
    v8f accLhi = {}, accLlo = {};
#endif

    for (int k0 = 0; k0 < kIn; k0 += 32) {
        float xa[16];
        {
            const float4 t0 = *(const float4*)(xrow + k0 + 8 * hs);
            const float4 t1 = *(const float4*)(xrow + k0 + 8 * hs + 4);
            const float4 t2 = *(const float4*)(xrow + k0 + 16 + 8 * hs);
            const float4 t3 = *(const float4*)(xrow + k0 + 16 + 8 * hs + 4);
            xa[0]=t0.x; xa[1]=t0.y; xa[2]=t0.z;  xa[3]=t0.w;
            xa[4]=t1.x; xa[5]=t1.y; xa[6]=t1.z;  xa[7]=t1.w;
            xa[8]=t2.x; xa[9]=t2.y; xa[10]=t2.z; xa[11]=t2.w;
            xa[12]=t3.x; xa[13]=t3.y; xa[14]=t3.z; xa[15]=t3.w;
        }
        v16h bM;
        {
            const float4 m0 = *(const float4*)(mrow + k0 + 8 * hs);
            const float4 m1 = *(const float4*)(mrow + k0 + 8 * hs + 4);
            const float4 m2 = *(const float4*)(mrow + k0 + 16 + 8 * hs);
            const float4 m3 = *(const float4*)(mrow + k0 + 16 + 8 * hs + 4);
            bM[0]=(_Float16)m0.x; bM[1]=(_Float16)m0.y; bM[2]=(_Float16)m0.z;  bM[3]=(_Float16)m0.w;
            bM[4]=(_Float16)m1.x; bM[5]=(_Float16)m1.y; bM[6]=(_Float16)m1.z;  bM[7]=(_Float16)m1.w;
            bM[8]=(_Float16)m2.x; bM[9]=(_Float16)m2.y; bM[10]=(_Float16)m2.z; bM[11]=(_Float16)m2.w;
            bM[12]=(_Float16)m3.x; bM[13]=(_Float16)m3.y; bM[14]=(_Float16)m3.z; bM[15]=(_Float16)m3.w;
        }
        v16h aZ, aS;
#pragma unroll
        for (int i = 0; i < 16; ++i) {
            const float xv = xa[i];
            aZ[i] = (xv != 0.0f) ? (_Float16)1.0f : (_Float16)0.0f;
            aS[i] = (xv <  0.0f) ? (_Float16)1.0f : (_Float16)0.0f;
        }
        accZ = __builtin_amdgcn_wmma_f32_16x16x32_f16(false, aZ, false, bM, (short)0, accZ, false, false);
        accS = __builtin_amdgcn_wmma_f32_16x16x32_f16(false, aS, false, bM, (short)0, accS, false, false);

#if HAVE_WMMA_F32X4
#pragma unroll
        for (int kk = 0; kk < 32; kk += 4) {
            const int kb = k0 + kk + 2 * hs;
            const float2 xv = *(const float2*)(xrow + kb);
            const float2 mv = *(const float2*)(mrow + kb);
            v2f aLg, bLg;
            aLg[0] = (xv.x != 0.0f) ? logf(fabsf(xv.x)) : 0.0f;
            aLg[1] = (xv.y != 0.0f) ? logf(fabsf(xv.y)) : 0.0f;
            bLg[0] = mv.x; bLg[1] = mv.y;
            accLg = __builtin_amdgcn_wmma_f32_16x16x4_f32(
                        false, aLg, false, bLg, (short)0, accLg, false, false);
        }
#else
        v16h aLhi, aLlo;
#pragma unroll
        for (int i = 0; i < 16; ++i) {
            const float xv = xa[i];
            const float L  = (xv != 0.0f) ? logf(fabsf(xv)) : 0.0f;
            const _Float16 hi = (_Float16)L;
            aLhi[i] = hi;
            aLlo[i] = (_Float16)(L - (float)hi);
        }
        accLhi = __builtin_amdgcn_wmma_f32_16x16x32_f16(false, aLhi, false, bM, (short)0, accLhi, false, false);
        accLlo = __builtin_amdgcn_wmma_f32_16x16x32_f16(false, aLlo, false, bM, (short)0, accLlo, false, false);
#endif
    }

#pragma unroll
    for (int v = 0; v < 8; ++v) {
#if HAVE_WMMA_F32X4
        const float sl = accLg[v];
#else
        const float sl = accLhi[v] + accLlo[v];
#endif
        const float cz = accZ[v];
        const int   sn = (int)(accS[v] + 0.5f);
        float val = expf(sl);
        if (sn & 1) val = -val;
        out[(size_t)(b0 + v + 8 * hs) * kOut + (o0 + row)] = (cz > 0.5f) ? val : 0.0f;
    }
}

// ---------------------------------------------------------------------------
extern "C" void kernel_launch(void* const* d_in, const int* in_sizes, int n_in,
                              void* d_out, int out_size, void* d_ws, size_t ws_size,
                              hipStream_t stream) {
    (void)in_sizes; (void)n_in; (void)out_size;
    const float* x    = (const float*)d_in[0];
    const float* mask = (const float*)d_in[1];
    float* out        = (float*)d_out;

    if (d_ws != nullptr && ws_size >= kWsNeed) {
        char* ws = (char*)d_ws;
        _Float16* lhi = (_Float16*)(ws + kOffLhi);
        _Float16* llo = (_Float16*)(ws + kOffLlo);
        _Float16* tct = (_Float16*)(ws + kOffT);
        _Float16* mh  = (_Float16*)(ws + kOffM);

        transform_x_kernel<<<dim3((kBatch * kIn) / (256 * 4)), dim3(256), 0, stream>>>(
            x, lhi, llo, tct);
        transform_mask_kernel<<<dim3((kOut * kIn) / (256 * 4)), dim3(256), 0, stream>>>(
            mask, mh);
        cn_gemm_kernel<<<dim3(kOut / 16, kBatch / (16 * kBG)), dim3(32), 0, stream>>>(
            lhi, llo, tct, mh, out);
    } else {
        cn_update_fused_kernel<<<dim3(kOut / 16, kBatch / 16), dim3(32), 0, stream>>>(
            x, mask, out);
    }
}
